// GptOssAttention_55405078118440
// MI455X (gfx1250) — matrix operand order repaired and checked by reading, not compiled, not run on hardware
//
#include <hip/hip_runtime.h>
#include <hip/hip_bf16.h>
#include <stdint.h>

typedef __bf16 bf16;
typedef __attribute__((ext_vector_type(4)))  __bf16 v4bf;
typedef __attribute__((ext_vector_type(8)))  __bf16 v8bf;
typedef __attribute__((ext_vector_type(16))) __bf16 v16bf;
typedef __attribute__((ext_vector_type(8)))  float  v8f;

#define S_LEN 2048
#define NQH   64
#define NKVH  8
#define HD    64
#define HID   2880
#define WIN   128

// ---------------------------------------------------------------------------
// WMMA fragment helpers (contiguous K order, k = 16*half + s, wave32).
// Strides are multiples of 8 bf16 so every chunk is 16B aligned and the LDS
// reads lower to ds_load_b128.
// ---------------------------------------------------------------------------

// A fragment 16x32 bf16 from row-major LDS tile [m][k]
__device__ __forceinline__ v16bf a_frag_lds(const bf16* base, int stride, int lane) {
  int m = lane & 15, kh = lane >> 4;
  const bf16* p = base + m * stride + kh * 16;
  v8bf lo = *(const v8bf*)(p);        // K = kh*16 + 0..7
  v8bf hi = *(const v8bf*)(p + 8);    // K = kh*16 + 8..15
  v16bf a;
#pragma unroll
  for (int i = 0; i < 8; ++i) { a[i] = lo[i]; a[i + 8] = hi[i]; }
  return a;
}

// B fragment 32x16 bf16 from LDS tile stored transposed [n][k]
__device__ __forceinline__ v16bf b_frag_lds_T(const bf16* base, int stride, int lane) {
  int n = lane & 15, kh = lane >> 4;
  const bf16* p = base + n * stride + kh * 16;   // K = kh*16 + 0..15
  v8bf lo = *(const v8bf*)(p);
  v8bf hi = *(const v8bf*)(p + 8);
  v16bf b;
#pragma unroll
  for (int i = 0; i < 8; ++i) { b[i] = lo[i]; b[i + 8] = hi[i]; }
  return b;
}

__device__ __forceinline__ v8f wmma_bf16(v16bf a, v16bf b, v8f c) {
  return __builtin_amdgcn_wmma_f32_16x16x32_bf16(false, a, false, b, (short)0, c,
                                                 false, false);
}

// ---------------------------------------------------------------------------
// GEMM: C[M,N] = A[M,K] * B[K,N] + bias[N], fp32 in/out, bf16 WMMA compute.
// Block tile 128x128, 256 threads (8 waves: 4 M x 2 N), each wave 32x64
// -> 8 WMMAs per K-step. K-step 32, double-buffered LDS, one barrier/step,
// next tile's global loads issued before the WMMA phase.
// ---------------------------------------------------------------------------
#define TM 128
#define TN 128
#define TK 32
#define LDK 40  // padded K-stride (bf16 elems), 80B rows keep 16B alignment

__global__ __launch_bounds__(256) void gemm_bf16_wmma(
    const float* __restrict__ A, const float* __restrict__ B,
    const float* __restrict__ bias, float* __restrict__ C,
    int M, int N, int K, int lda, int ldb, int ldc) {
  __shared__ bf16 As[2][TM][LDK];   // row-major (m, k)
  __shared__ bf16 Bs[2][TN][LDK];   // transposed (n, k)

  const int tid  = threadIdx.x;
  const int lane = tid & 31;
  const int wave = tid >> 5;
  const int wm   = wave & 3;      // M sub-tile (32 rows)
  const int wn   = wave >> 2;     // N sub-tile (64 cols)
  const int m0   = blockIdx.x * TM;
  const int n0   = blockIdx.y * TN;
  const int nsteps = K / TK;

  v8f acc[2][4];
#pragma unroll
  for (int u = 0; u < 2; ++u)
#pragma unroll
    for (int t = 0; t < 4; ++t)
#pragma unroll
      for (int i = 0; i < 8; ++i) acc[u][t][i] = 0.0f;

  float4 areg[4];     // A: 4 quads along K per thread
  float  breg[4][4];  // B: 4 quads (K-strided) per thread

  // Issue all global loads for tile at k0 (one clause, no intermediate waits)
  auto load_tiles = [&](int k0) {
#pragma unroll
    for (int j = 0; j < 4; ++j) {           // A: 128x32 as 1024 float4
      int i = tid + j * 256;
      int m = i >> 3, kq = (i & 7) * 4;
      areg[j] = *(const float4*)&A[(size_t)(m0 + m) * lda + k0 + kq];
    }
#pragma unroll
    for (int j = 0; j < 4; ++j) {           // B: 32x128, 4 K-strided scalars
      int i = tid + j * 256;
      int n = i & 127, kq = (i >> 7) * 4;
      int gn  = n0 + n;
      int gnc = gn < N ? gn : N - 1;        // clamp (no OOB), zeroed at store
#pragma unroll
      for (int jj = 0; jj < 4; ++jj)
        breg[j][jj] = B[(size_t)(k0 + kq + jj) * ldb + gnc];
    }
  };

  // Convert to bf16 and store staged registers into LDS buffer `buf`
  auto store_tiles = [&](int buf) {
#pragma unroll
    for (int j = 0; j < 4; ++j) {
      int i = tid + j * 256;
      int m = i >> 3, kq = (i & 7) * 4;
      v4bf pk;
      pk[0] = (bf16)areg[j].x; pk[1] = (bf16)areg[j].y;
      pk[2] = (bf16)areg[j].z; pk[3] = (bf16)areg[j].w;
      *(v4bf*)&As[buf][m][kq] = pk;
    }
#pragma unroll
    for (int j = 0; j < 4; ++j) {
      int i = tid + j * 256;
      int n = i & 127, kq = (i >> 7) * 4;
      bool ok = (n0 + n) < N;
      v4bf pk;
#pragma unroll
      for (int jj = 0; jj < 4; ++jj) pk[jj] = (bf16)(ok ? breg[j][jj] : 0.0f);
      *(v4bf*)&Bs[buf][n][kq] = pk;
    }
  };

  load_tiles(0);
  store_tiles(0);
  __syncthreads();

  for (int step = 0; step < nsteps; ++step) {
    const bool has_next = (step + 1) < nsteps;
    if (has_next) load_tiles((step + 1) * TK);   // overlap with WMMAs below

    const bf16* as = &As[step & 1][0][0];
    const bf16* bs = &Bs[step & 1][0][0];
    v16bf a0 = a_frag_lds(as + (wm * 32) * LDK, LDK, lane);
    v16bf a1 = a_frag_lds(as + (wm * 32 + 16) * LDK, LDK, lane);
#pragma unroll
    for (int t = 0; t < 4; ++t) {
      v16bf b = b_frag_lds_T(bs + (wn * 64 + t * 16) * LDK, LDK, lane);
      acc[0][t] = wmma_bf16(a0, b, acc[0][t]);
      acc[1][t] = wmma_bf16(a1, b, acc[1][t]);
    }
    if (has_next) {
      store_tiles((step + 1) & 1);
      __syncthreads();
    }
  }

  const int kh = lane >> 4;
#pragma unroll
  for (int u = 0; u < 2; ++u) {
#pragma unroll
    for (int t = 0; t < 4; ++t) {
      int n = n0 + wn * 64 + t * 16 + (lane & 15);
      if (n < N) {
        float bv = bias[n];
#pragma unroll
        for (int r = 0; r < 8; ++r) {
          int m = m0 + wm * 32 + u * 16 + r + 8 * kh;
          C[(size_t)m * ldc + n] = acc[u][t][r] + bv;
        }
      }
    }
  }
}

// ---------------------------------------------------------------------------
// RoPE (split-half rotate), in place on (s, n_heads, 64) fp32
// ---------------------------------------------------------------------------
__global__ __launch_bounds__(256) void rope_kernel(float* __restrict__ x,
                                                   int n_heads, int total) {
  int idx = blockIdx.x * blockDim.x + threadIdx.x;
  if (idx >= total) return;
  int j    = idx & 31;
  int ph   = idx >> 5;
  int head = ph % n_heads;
  int pos  = ph / n_heads;
  float inv_freq = __powf(10000.0f, -(float)(2 * j) * (1.0f / 64.0f));
  float ang = (float)pos * inv_freq;
  float sn, cs;
  __sincosf(ang, &sn, &cs);
  float* base = x + ((size_t)pos * n_heads + head) * HD;
  float x1 = base[j], x2 = base[j + 32];
  base[j]      = x1 * cs - x2 * sn;
  base[j + 32] = x2 * cs + x1 * sn;
}

// ---------------------------------------------------------------------------
// Sliding-window attention with sink, GQA 8:1, flash-style online softmax.
// Grid: (S/64, NQH). Block: 128 threads = 4 waves, each wave = 16 queries.
// Uniform 6 key-blocks of 32 covering [Q0-128, Q0+63].
// ---------------------------------------------------------------------------
__global__ __launch_bounds__(128) void attn_kernel(
    const float* __restrict__ qw,   // (s, 64, 64)
    const float* __restrict__ kw,   // (s, 8, 64)
    const float* __restrict__ vw,   // (s, 8, 64)
    const float* __restrict__ sinks,
    float* __restrict__ ow) {       // (s, 64, 64)
  __shared__ bf16 Pbuf[4][16 * LDK];

  const int lane = threadIdx.x & 31;
  const int wave = threadIdx.x >> 5;
  const int h    = blockIdx.y;
  const int kvh  = h >> 3;
  const int Q0   = blockIdx.x * 64;
  const int q0   = Q0 + wave * 16;
  const int mrow = lane & 15;
  const int kh   = lane >> 4;
  const float inv_norm = 0.125f;  // 1/sqrt(64)

  // Preload Q fragments (two 32-dim chunks), vectorized fp32 -> bf16
  v16bf qa[2];
#pragma unroll
  for (int c = 0; c < 2; ++c) {
    const float* p = qw + ((size_t)(q0 + mrow) * NQH + h) * HD + c * 32 + kh * 16;
    float4 f0 = *(const float4*)(p);         // K = kh*16 + 0..3
    float4 f1 = *(const float4*)(p + 4);     // K = kh*16 + 4..7
    float4 f2 = *(const float4*)(p + 8);     // K = kh*16 + 8..11
    float4 f3 = *(const float4*)(p + 12);
    v16bf a;
    a[0]  = (bf16)f0.x; a[1]  = (bf16)f0.y; a[2]  = (bf16)f0.z; a[3]  = (bf16)f0.w;
    a[4]  = (bf16)f1.x; a[5]  = (bf16)f1.y; a[6]  = (bf16)f1.z; a[7]  = (bf16)f1.w;
    a[8]  = (bf16)f2.x; a[9]  = (bf16)f2.y; a[10] = (bf16)f2.z; a[11] = (bf16)f2.w;
    a[12] = (bf16)f3.x; a[13] = (bf16)f3.y; a[14] = (bf16)f3.z; a[15] = (bf16)f3.w;
    qa[c] = a;
  }

  // Online-softmax state; sink logit folded into the init (m=sink, l=1).
  float snk = sinks[h];
  float m_r[8], l_r[8];
#pragma unroll
  for (int r = 0; r < 8; ++r) { m_r[r] = snk; l_r[r] = 1.0f; }
  v8f o[4];
#pragma unroll
  for (int t = 0; t < 4; ++t)
#pragma unroll
    for (int i = 0; i < 8; ++i) o[t][i] = 0.0f;

  for (int kb = 0; kb < 6; ++kb) {
    const int key0 = Q0 - WIN + kb * 32;        // multiple of 32
    if (key0 < 0) continue;                     // fully OOB (block-uniform)
    if (key0 > q0 + 15) continue;               // fully beyond causal (wave-uniform)

    // --- scores: S(16q x 32keys) = Q (16x64) * K^T, two 16-key column tiles
    v8f s[2];
#pragma unroll
    for (int t = 0; t < 2; ++t) {
#pragma unroll
      for (int i = 0; i < 8; ++i) s[t][i] = 0.0f;
#pragma unroll
      for (int c = 0; c < 2; ++c) {
        // B(kk,n) = K[key0 + t*16 + n][c*32 + kk], 16 contiguous floats
        const float* p = kw + ((size_t)(key0 + t * 16 + mrow) * NKVH + kvh) * HD +
                         c * 32 + kh * 16;
        v16bf b;
#pragma unroll
        for (int j = 0; j < 4; ++j) {
          float4 f = *(const float4*)(p + 4 * j);
          b[4 * j]     = (bf16)f.x;
          b[4 * j + 1] = (bf16)f.y;
          b[4 * j + 2] = (bf16)f.z;
          b[4 * j + 3] = (bf16)f.w;
        }
        s[t] = wmma_bf16(qa[c], b, s[t]);
      }
    }

    // --- mask + online softmax (row = r + 8*kh, per-half 16-lane reductions)
    float fac[8], p0[8], p1[8];
#pragma unroll
    for (int r = 0; r < 8; ++r) {
      int qrow = q0 + r + 8 * kh;
      int ka  = key0 + mrow;
      int kb2 = key0 + 16 + mrow;
      float l0 = (ka  <= qrow && ka  >= qrow - WIN) ? s[0][r] * inv_norm : -1e9f;
      float l1 = (kb2 <= qrow && kb2 >= qrow - WIN) ? s[1][r] * inv_norm : -1e9f;
      float tmax = fmaxf(l0, l1);
#pragma unroll
      for (int msk = 8; msk >= 1; msk >>= 1)
        tmax = fmaxf(tmax, __shfl_xor(tmax, msk, 32));
      float nm = fmaxf(m_r[r], tmax);
      float f  = __expf(m_r[r] - nm);
      float e0 = __expf(l0 - nm);
      float e1 = __expf(l1 - nm);
      float ps = e0 + e1;
#pragma unroll
      for (int msk = 8; msk >= 1; msk >>= 1)
        ps += __shfl_xor(ps, msk, 32);
      l_r[r] = l_r[r] * f + ps;
      m_r[r] = nm;
      fac[r] = f; p0[r] = e0; p1[r] = e1;
    }
#pragma unroll
    for (int t = 0; t < 4; ++t)
#pragma unroll
      for (int r = 0; r < 8; ++r) o[t][r] *= fac[r];

    // --- P (16x32) -> per-wave LDS as bf16, then re-gather as A fragment
    bf16* pb = &Pbuf[wave][0];
#pragma unroll
    for (int r = 0; r < 8; ++r) {
      int row = r + 8 * kh;
      pb[row * LDK + mrow]      = (bf16)p0[r];
      pb[row * LDK + 16 + mrow] = (bf16)p1[r];
    }
    asm volatile("s_wait_dscnt 0" ::: "memory");  // same-wave LDS RAW
    v16bf pa = a_frag_lds(pb, LDK, lane);

    // --- O += P * V   (V tile: 32 keys x 64 dims, 4 N-tiles)
#pragma unroll
    for (int t = 0; t < 4; ++t) {
      const float* p = vw + ((size_t)(key0 + kh * 16) * NKVH + kvh) * HD +
                       t * 16 + mrow;
      v16bf b;
#pragma unroll
      for (int e = 0; e < 16; ++e) b[e] = (bf16)p[(size_t)e * (NKVH * HD)];
      o[t] = wmma_bf16(pa, b, o[t]);
    }
  }

  // --- normalize and store (s, 64, 64)
#pragma unroll
  for (int t = 0; t < 4; ++t) {
#pragma unroll
    for (int r = 0; r < 8; ++r) {
      int row = q0 + r + 8 * kh;
      ow[((size_t)row * NQH + h) * HD + t * 16 + mrow] = o[t][r] / l_r[r];
    }
  }
}

// ---------------------------------------------------------------------------
// Launch
// ---------------------------------------------------------------------------
extern "C" void kernel_launch(void* const* d_in, const int* in_sizes, int n_in,
                              void* d_out, int out_size, void* d_ws, size_t ws_size,
                              hipStream_t stream) {
  const float* hs    = (const float*)d_in[0];   // (1, 2048, 2880)
  const float* Wq    = (const float*)d_in[1];   // (2880, 64, 64)
  const float* bq    = (const float*)d_in[2];   // (64, 64)
  const float* Wk    = (const float*)d_in[3];   // (2880, 8, 64)
  const float* bk    = (const float*)d_in[4];
  const float* Wv    = (const float*)d_in[5];
  const float* bv    = (const float*)d_in[6];
  const float* Wo    = (const float*)d_in[7];   // (64, 64, 2880)
  const float* bo    = (const float*)d_in[8];   // (2880,)
  const float* sinks = (const float*)d_in[9];   // (64,)
  float* out = (float*)d_out;                   // (1, 2048, 2880)

  float* q_ws = (float*)d_ws;                              // 2048*4096
  float* k_ws = q_ws + (size_t)S_LEN * NQH * HD;           // 2048*512
  float* v_ws = k_ws + (size_t)S_LEN * NKVH * HD;          // 2048*512
  float* a_ws = v_ws + (size_t)S_LEN * NKVH * HD;          // 2048*4096

  const dim3 blk(256);
  const int QN  = NQH * HD;   // 4096
  const int KVN = NKVH * HD;  // 512

  // QKV projections
  gemm_bf16_wmma<<<dim3(S_LEN / TM, QN / TN), blk, 0, stream>>>(
      hs, Wq, bq, q_ws, S_LEN, QN, HID, HID, QN, QN);
  gemm_bf16_wmma<<<dim3(S_LEN / TM, KVN / TN), blk, 0, stream>>>(
      hs, Wk, bk, k_ws, S_LEN, KVN, HID, HID, KVN, KVN);
  gemm_bf16_wmma<<<dim3(S_LEN / TM, KVN / TN), blk, 0, stream>>>(
      hs, Wv, bv, v_ws, S_LEN, KVN, HID, HID, KVN, KVN);

  // RoPE on q and k
  int tq = S_LEN * NQH * 32;
  rope_kernel<<<(tq + 255) / 256, 256, 0, stream>>>(q_ws, NQH, tq);
  int tk = S_LEN * NKVH * 32;
  rope_kernel<<<(tk + 255) / 256, 256, 0, stream>>>(k_ws, NKVH, tk);

  // Sliding-window attention with sink
  attn_kernel<<<dim3(S_LEN / 64, NQH), dim3(128), 0, stream>>>(
      q_ws, k_ws, v_ws, sinks, a_ws);

  // Output projection
  gemm_bf16_wmma<<<dim3(S_LEN / TM, (HID + TN - 1) / TN), blk, 0, stream>>>(
      a_ws, Wo, bo, out, S_LEN, HID, QN, QN, HID, HID);
}
